// PerformerSeperator_35536559407789
// MI455X (gfx1250) — compile-verified, matrix operand order checked by reading, not run on hardware
//
#include <hip/hip_runtime.h>
#include <hip/hip_bf16.h>
#include <math.h>

// ---------------- problem constants ----------------
#define B_   32
#define F_   256
#define T_   512
#define DIM_ 512
#define L_   6
#define H_   8
#define M_   256
#define DH_  64
#define FFD_ 2048
#define NM_  4

#define DN_    0.35355339059327373f  // 64^-0.25
#define RATIO_ 0.0625f               // 256^-0.5
#define FEPS_  1e-4f

typedef __bf16 bf16_t;
typedef bf16_t v16bf __attribute__((ext_vector_type(16)));
typedef bf16_t v8bf  __attribute__((ext_vector_type(8)));
typedef float  v8f   __attribute__((ext_vector_type(8)));

__device__ __forceinline__ bf16_t f2bf(float f) {
  unsigned u = __builtin_bit_cast(unsigned, f);
  unsigned r = u + 0x7FFFu + ((u >> 16) & 1u);   // round-to-nearest-even
  unsigned short h = (unsigned short)(r >> 16);
  return __builtin_bit_cast(bf16_t, h);
}

__device__ __forceinline__ float wred_sum(float v) {
#pragma unroll
  for (int o = 16; o > 0; o >>= 1) v += __shfl_xor(v, o, 32);
  return v;
}
__device__ __forceinline__ float wred_max(float v) {
#pragma unroll
  for (int o = 16; o > 0; o >>= 1) v = fmaxf(v, __shfl_xor(v, o, 32));
  return v;
}
__device__ __forceinline__ float gelu_f(float x) {
  float x3 = x * x * x;
  return 0.5f * x * (1.f + tanhf(0.79788456080286535588f * (x + 0.044715f * x3)));
}

// ---------------- generic batched strided WMMA GEMM ----------------
// C[z][m][n] = op( sum_k A[z][m][k]*scaleA * B[z][k][n] + bias[n] (+ R[z][m][n]) )
// batch offset for X in {A,B,C}: (z/divX)*sX1 + (z%divX)*sX2
//
// Block tile 256x128 (8 waves, each owning a 64x64 sub-tile -> 16 WMMA / K-step).
// LDS row pitch 40 bf16 = 80 B: rows stay 16B-aligned (b128 fragment reads) and
// 20-bank stride makes the 16-lane column gather conflict-free.
#define BM 256
#define BN 128
#define BK 32
#define BKP 40

// fragment = two contiguous 16B runs per lane:
//   elems 0..7  = K[hl*8 .. hl*8+7], elems 8..15 = K[16+hl*8 .. 23+hl*8]
__device__ __forceinline__ v16bf load_frag(const bf16_t* rowbase, int hl) {
  const v8bf lo = *(const v8bf*)(rowbase + hl * 8);
  const v8bf hi = *(const v8bf*)(rowbase + 16 + hl * 8);
  return __builtin_shufflevector(lo, hi, 0, 1, 2, 3, 4, 5, 6, 7,
                                 8, 9, 10, 11, 12, 13, 14, 15);
}

__global__ __launch_bounds__(256) void gemm_wmma_bf16(
    float* __restrict__ C, const float* __restrict__ A, const float* __restrict__ Bm,
    const float* __restrict__ bias, const float* __restrict__ Rsd,
    int Msz, int Nsz, int Ksz,
    long long rsA, long long csA, long long sA1, long long sA2, int divA,
    long long rsB, long long csB, long long sB1, long long sB2, int divB,
    long long rsC, long long csC, long long sC1, long long sC2, int divC,
    long long sR1, long long sR2,
    float scaleA, int op)
{
  __shared__ __align__(16) bf16_t shA[BM * BKP];   // 20 KB
  __shared__ __align__(16) bf16_t shB[BN * BKP];   // 10 KB, transposed: shB[n][k]

  const int z = blockIdx.z;
  const float* Ab = A  + (long long)(z / divA) * sA1 + (long long)(z % divA) * sA2;
  const float* Bb = Bm + (long long)(z / divB) * sB1 + (long long)(z % divB) * sB2;
  float*       Cb = C  + (long long)(z / divC) * sC1 + (long long)(z % divC) * sC2;
  const float* Rb = Rsd ? (Rsd + (long long)(z / divC) * sR1 + (long long)(z % divC) * sR2)
                        : nullptr;

  const int m0 = blockIdx.y * BM;
  const int n0 = blockIdx.x * BN;
  const int tid  = threadIdx.x;
  const int lane = tid & 31;
  const int wv   = tid >> 5;   // 0..7
  const int wr   = wv >> 1;    // 0..3 : 64-row band
  const int wc   = wv & 1;     // 0..1 : 64-col band
  const int hl   = lane >> 4;  // lane half
  const int rr   = lane & 15;

  v8f acc[4][4];
#pragma unroll
  for (int ti = 0; ti < 4; ++ti)
#pragma unroll
    for (int tj = 0; tj < 4; ++tj)
#pragma unroll
      for (int jj = 0; jj < 8; ++jj) acc[ti][tj][jj] = 0.f;

  for (int k0 = 0; k0 < Ksz; k0 += BK) {
    // stage A tile (BM x BK), convert f32 -> bf16
#pragma unroll
    for (int i = tid; i < BM * BK; i += 256) {
      int m = i >> 5, k = i & 31;
      int gm = m0 + m;
      float v = 0.f;
      if (gm < Msz) v = Ab[(long long)gm * rsA + (long long)(k0 + k) * csA] * scaleA;
      shA[m * BKP + k] = f2bf(v);
    }
    // stage B tile transposed (shB[n][k])
#pragma unroll
    for (int i = tid; i < BN * BK; i += 256) {
      int n = i & 127, k = i >> 7;
      int gn = n0 + n;
      float v = 0.f;
      if (gn < Nsz) v = Bb[(long long)(k0 + k) * rsB + (long long)gn * csB];
      shB[n * BKP + k] = f2bf(v);
    }
    // prefetch next K tiles (global_prefetch_b8)
    if (k0 + BK < Ksz) {
      int pm = m0 + (tid >> 5) * 8;
      if (pm < Msz)
        __builtin_prefetch(&Ab[(long long)pm * rsA + (long long)(k0 + BK + (tid & 31)) * csA], 0, 0);
      int pn = n0 + (tid & 127);
      if (pn < Nsz)
        __builtin_prefetch(&Bb[(long long)(k0 + BK + (tid >> 7)) * rsB + (long long)pn * csB], 0, 0);
    }
    __syncthreads();

    // fragments: 2 x ds_load_b128 each
    v16bf af[4], bfq[4];
#pragma unroll
    for (int ti = 0; ti < 4; ++ti) {
      int row = wr * 64 + ti * 16 + rr;
      af[ti] = load_frag(&shA[row * BKP], hl);
    }
#pragma unroll
    for (int tj = 0; tj < 4; ++tj) {
      int col = wc * 64 + tj * 16 + rr;
      bfq[tj] = load_frag(&shB[col * BKP], hl);
    }
#pragma unroll
    for (int ti = 0; ti < 4; ++ti)
#pragma unroll
      for (int tj = 0; tj < 4; ++tj)
        acc[ti][tj] = __builtin_amdgcn_wmma_f32_16x16x32_bf16(
            false, af[ti], false, bfq[tj], (short)0, acc[ti][tj], false, false);
    __syncthreads();
  }

  // epilogue: bias / gelu / residual, strided store
#pragma unroll
  for (int ti = 0; ti < 4; ++ti)
#pragma unroll
    for (int tj = 0; tj < 4; ++tj) {
      int n = n0 + wc * 64 + tj * 16 + rr;
      if (n < Nsz) {
        float bv = bias ? bias[n] : 0.f;
#pragma unroll
        for (int jj = 0; jj < 8; ++jj) {
          int m = m0 + wr * 64 + ti * 16 + hl * 8 + jj;
          if (m < Msz) {
            float v = acc[ti][tj][jj] + bv;
            if (op == 1) v = gelu_f(v);
            long long ci = (long long)m * rsC + (long long)n * csC;
            if (op == 2) v += Rb[ci];
            Cb[ci] = v;
          }
        }
      }
    }
}

// ---------------- elementwise / reduction kernels ----------------

// one wave per row of DIM_=512
__global__ __launch_bounds__(256) void ln_kernel(float* __restrict__ out,
    const float* __restrict__ in, const float* __restrict__ g, const float* __restrict__ bt)
{
  int lane = threadIdx.x & 31, wv = threadIdx.x >> 5;
  long long row = (long long)blockIdx.x * 8 + wv;
  const float* x = in + row * DIM_;
  float vals[16], s = 0.f;
#pragma unroll
  for (int i = 0; i < 16; ++i) { vals[i] = x[lane + 32 * i]; s += vals[i]; }
  float mu = wred_sum(s) * (1.f / DIM_);
  float v = 0.f;
#pragma unroll
  for (int i = 0; i < 16; ++i) { float d = vals[i] - mu; v += d * d; }
  float inv = rsqrtf(wred_sum(v) * (1.f / DIM_) + 1e-5f);
  float* o = out + row * DIM_;
#pragma unroll
  for (int i = 0; i < 16; ++i) {
    int c = lane + 32 * i;
    o[c] = (vals[i] - mu) * inv * g[c] + bt[c];
  }
}

// per (b,h): global max of kp over (T,M)
__global__ __launch_bounds__(256) void kmax_kernel(float* __restrict__ kmax,
                                                   const float* __restrict__ kp)
{
  __shared__ float red[256];
  const int z = blockIdx.x;
  const float* p = kp + (long long)z * T_ * M_;
  float m = -3.402823466e38f;
  for (int i = threadIdx.x; i < T_ * M_; i += 256) m = fmaxf(m, p[i]);
  red[threadIdx.x] = m;
  __syncthreads();
  for (int s = 128; s > 0; s >>= 1) {
    if (threadIdx.x < s) red[threadIdx.x] = fmaxf(red[threadIdx.x], red[threadIdx.x + s]);
    __syncthreads();
  }
  if (threadIdx.x == 0) kmax[z] = red[0];
}

// FAVOR feature map, in place on phi (one wave per (b,h,t) row of M_=256).
// sq = 0.5*sum_d (x_d*dn)^2 computed from qkv; stabilizer = row max (q) or global max (k).
__global__ __launch_bounds__(256) void phi_kernel(float* __restrict__ phi,
    const float* __restrict__ qkv, const float* __restrict__ gmax, int koff, int useg)
{
  int lane = threadIdx.x & 31, wv = threadIdx.x >> 5;
  long long row = (long long)blockIdx.x * 8 + wv;   // z*T + t
  int z = (int)(row >> 9), t = (int)(row & 511);
  int b = z >> 3, h = z & 7;
  const float* qr = qkv + ((long long)b * T_ + t) * (3 * DIM_) + koff + h * DH_;
  float sq = 0.f;
#pragma unroll
  for (int i = 0; i < 2; ++i) { float v = qr[lane + 32 * i] * DN_; sq += v * v; }
  sq = 0.5f * wred_sum(sq);
  float* pr = phi + row * M_;
  float vals[8];
#pragma unroll
  for (int i = 0; i < 8; ++i) vals[i] = pr[lane + 32 * i];
  float mx;
  if (useg) {
    mx = gmax[z];
  } else {
    mx = vals[0];
#pragma unroll
    for (int i = 1; i < 8; ++i) mx = fmaxf(mx, vals[i]);
    mx = wred_max(mx);
  }
#pragma unroll
  for (int i = 0; i < 8; ++i)
    pr[lane + 32 * i] = RATIO_ * (expf(vals[i] - sq - mx) + FEPS_);
}

// ksum[z][m] = sum_t kphi[z][t][m]
__global__ __launch_bounds__(256) void ksum_kernel(float* __restrict__ ksum,
                                                   const float* __restrict__ kphi)
{
  int z = blockIdx.x, m = threadIdx.x;
  const float* p = kphi + (long long)z * T_ * M_ + m;
  float s = 0.f;
  for (int t = 0; t < T_; ++t) s += p[(long long)t * M_];
  ksum[(long long)z * M_ + m] = s;
}

// den[z][t] = dot(qphi[z][t][:], ksum[z][:]) — one wave per row
__global__ __launch_bounds__(256) void den_kernel(float* __restrict__ den,
    const float* __restrict__ qphi, const float* __restrict__ ksum)
{
  int lane = threadIdx.x & 31, wv = threadIdx.x >> 5;
  long long row = (long long)blockIdx.x * 8 + wv;   // z*T + t
  int z = (int)(row >> 9);
  const float* q = qphi + row * M_;
  const float* k = ksum + (long long)z * M_;
  float s = 0.f;
#pragma unroll
  for (int i = 0; i < 8; ++i) s += q[lane + 32 * i] * k[lane + 32 * i];
  s = wred_sum(s);
  if (lane == 0) den[row] = s;
}

// ob laid out (B,T,H,DH); divide by den[b,h,t]+1e-6
__global__ __launch_bounds__(256) void div_kernel(float* __restrict__ ob,
                                                  const float* __restrict__ den)
{
  long long idx = (long long)blockIdx.x * 256 + threadIdx.x;
  int h = (int)((idx >> 6) & 7);
  int t = (int)((idx >> 9) & 511);
  int b = (int)(idx >> 18);
  float d = den[((long long)(b * 8 + h)) * 512 + t] + 1e-6f;
  ob[idx] = ob[idx] / d;
}

// out[b][nm][t] = sigmoid(X[b][t][:] . mask_w[:,nm] + mask_b[nm])
__global__ __launch_bounds__(256) void mask_kernel(float* __restrict__ out,
    const float* __restrict__ X, const float* __restrict__ mw, const float* __restrict__ mb)
{
  int idx = blockIdx.x * 256 + threadIdx.x;   // (b*NM + nm)*T + t
  int t  = idx & (T_ - 1);
  int nm = (idx >> 9) & (NM_ - 1);
  int b  = idx >> 11;
  const float* xr = X + ((long long)b * T_ + t) * DIM_;
  float s = mb[nm];
  for (int d = 0; d < DIM_; ++d) s += xr[d] * mw[d * NM_ + nm];
  out[idx] = 1.f / (1.f + expf(-s));
}

// ---------------- host launcher ----------------
static void launch_gemm(hipStream_t s, float* C, const float* A, const float* Bm,
                        const float* bias, const float* Rsd,
                        int Msz, int Nsz, int Ksz,
                        long long rsA, long long csA, long long sA1, long long sA2, int divA,
                        long long rsB, long long csB, long long sB1, long long sB2, int divB,
                        long long rsC, long long csC, long long sC1, long long sC2, int divC,
                        long long sR1, long long sR2, float scaleA, int op, int batches)
{
  dim3 grid((Nsz + BN - 1) / BN, (Msz + BM - 1) / BM, batches);
  gemm_wmma_bf16<<<grid, dim3(256), 0, s>>>(C, A, Bm, bias, Rsd, Msz, Nsz, Ksz,
      rsA, csA, sA1, sA2, divA, rsB, csB, sB1, sB2, divB,
      rsC, csC, sC1, sC2, divC, sR1, sR2, scaleA, op);
}

extern "C" void kernel_launch(void* const* d_in, const int* in_sizes, int n_in,
                              void* d_out, int out_size, void* d_ws, size_t ws_size,
                              hipStream_t stream)
{
  (void)in_sizes; (void)n_in; (void)out_size; (void)ws_size;
  const float* mel      = (const float*)d_in[0];
  const float* to_emb_w = (const float*)d_in[1];
  const float* to_emb_b = (const float*)d_in[2];
  const float* pos_emb  = (const float*)d_in[3];
  const float* proj     = (const float*)d_in[4];
  const float* ln1_g    = (const float*)d_in[5];
  const float* ln1_b    = (const float*)d_in[6];
  const float* qkv_w    = (const float*)d_in[7];
  const float* qkv_b    = (const float*)d_in[8];
  const float* out_w    = (const float*)d_in[9];
  const float* out_b    = (const float*)d_in[10];
  const float* ln2_g    = (const float*)d_in[11];
  const float* ln2_b    = (const float*)d_in[12];
  const float* ff1_w    = (const float*)d_in[13];
  const float* ff1_b    = (const float*)d_in[14];
  const float* ff2_w    = (const float*)d_in[15];
  const float* ff2_b    = (const float*)d_in[16];
  const float* mask_w   = (const float*)d_in[17];
  const float* mask_b   = (const float*)d_in[18];

  char* ws = (char*)d_ws;
  size_t off = 0;
  auto wsalloc = [&](size_t bytes) -> float* {
    float* p = (float*)(ws + off);
    off += (bytes + 255) & ~(size_t)255;
    return p;
  };
  float* X    = wsalloc((size_t)B_ * T_ * DIM_ * 4);       // 32 MB residual stream
  float* Hb   = wsalloc((size_t)B_ * T_ * DIM_ * 4);       // 32 MB post-LN
  float* QKV  = wsalloc((size_t)B_ * T_ * 3 * DIM_ * 4);   // 96 MB
  float* QP   = wsalloc((size_t)B_ * H_ * T_ * M_ * 4);    // 128 MB (also FF1/GELU buffer)
  float* KP   = wsalloc((size_t)B_ * H_ * T_ * M_ * 4);    // 128 MB
  float* KV   = wsalloc((size_t)B_ * H_ * M_ * DH_ * 4);   // 16 MB
  float* OB   = wsalloc((size_t)B_ * T_ * DIM_ * 4);       // 32 MB attn out (B,T,H,DH)
  float* KSUM = wsalloc((size_t)B_ * H_ * M_ * 4);
  float* DEN  = wsalloc((size_t)B_ * H_ * T_ * 4);
  float* KMX  = wsalloc((size_t)B_ * H_ * 4);

  const long long RQKV = 3 * DIM_;                // 1536
  const long long sAb  = (long long)T_ * RQKV;    // per-batch stride in qkv

  // x = einsum('bft,fd', mel, to_emb_w) + to_emb_b + pos_emb   (batched over b; A read transposed)
  launch_gemm(stream, X, mel, to_emb_w, to_emb_b, pos_emb,
              T_, DIM_, F_,
              1, T_, (long long)F_ * T_, 0, 1,
              DIM_, 1, 0, 0, 1,
              DIM_, 1, (long long)T_ * DIM_, 0, 1,
              0, 0, 1.f, 2, B_);

  for (int i = 0; i < L_; ++i) {
    const float* pj = proj + (size_t)i * M_ * DH_;

    ln_kernel<<<2048, 256, 0, stream>>>(Hb, X, ln1_g + (size_t)i * DIM_, ln1_b + (size_t)i * DIM_);

    // qkv = h @ qkv_w + qkv_b
    launch_gemm(stream, QKV, Hb, qkv_w + (size_t)i * DIM_ * RQKV,
                qkv_b + (size_t)i * RQKV, nullptr,
                B_ * T_, (int)RQKV, DIM_,
                DIM_, 1, 0, 0, 1,
                RQKV, 1, 0, 0, 1,
                RQKV, 1, 0, 0, 1,
                0, 0, 1.f, 0, 1);

    // qp = (q*dn) @ projT   per (b,h);  A reads head-sliced qkv directly
    launch_gemm(stream, QP, QKV, pj, nullptr, nullptr,
                T_, M_, DH_,
                RQKV, 1, sAb, DH_, H_,
                1, DH_, 0, 0, 1,
                M_, 1, (long long)T_ * M_, 0, 1,
                0, 0, DN_, 0, B_ * H_);
    // kp
    launch_gemm(stream, KP, QKV + DIM_, pj, nullptr, nullptr,
                T_, M_, DH_,
                RQKV, 1, sAb, DH_, H_,
                1, DH_, 0, 0, 1,
                M_, 1, (long long)T_ * M_, 0, 1,
                0, 0, DN_, 0, B_ * H_);

    kmax_kernel<<<B_ * H_, 256, 0, stream>>>(KMX, KP);
    phi_kernel<<<16384, 256, 0, stream>>>(QP, QKV, KMX, 0, 0);      // q: per-row max
    phi_kernel<<<16384, 256, 0, stream>>>(KP, QKV, KMX, DIM_, 1);   // k: global max

    // kv[m][d] = sum_t kphi[t][m] * v[t][d]
    launch_gemm(stream, KV, KP, QKV + 2 * DIM_, nullptr, nullptr,
                M_, DH_, T_,
                1, M_, (long long)T_ * M_, 0, 1,
                RQKV, 1, sAb, DH_, H_,
                DH_, 1, (long long)M_ * DH_, 0, 1,
                0, 0, 1.f, 0, B_ * H_);

    ksum_kernel<<<B_ * H_, M_, 0, stream>>>(KSUM, KP);
    den_kernel<<<16384, 256, 0, stream>>>(DEN, QP, KSUM);

    // out_num = qphi @ kv   -> OB laid out (B,T,H,DH)
    launch_gemm(stream, OB, QP, KV, nullptr, nullptr,
                T_, DH_, M_,
                M_, 1, (long long)T_ * M_, 0, 1,
                DH_, 1, (long long)M_ * DH_, 0, 1,
                DIM_, 1, (long long)T_ * DIM_, DH_, H_,
                0, 0, 1.f, 0, B_ * H_);
    div_kernel<<<65536, 256, 0, stream>>>(OB, DEN);

    // x = x + o @ out_w + out_b   (in-place residual)
    launch_gemm(stream, X, OB, out_w + (size_t)i * DIM_ * DIM_,
                out_b + (size_t)i * DIM_, X,
                B_ * T_, DIM_, DIM_,
                DIM_, 1, 0, 0, 1,
                DIM_, 1, 0, 0, 1,
                DIM_, 1, 0, 0, 1,
                0, 0, 1.f, 2, 1);

    ln_kernel<<<2048, 256, 0, stream>>>(Hb, X, ln2_g + (size_t)i * DIM_, ln2_b + (size_t)i * DIM_);

    // ff1 + gelu -> QP (reused, exactly 128 MB)
    launch_gemm(stream, QP, Hb, ff1_w + (size_t)i * DIM_ * FFD_,
                ff1_b + (size_t)i * FFD_, nullptr,
                B_ * T_, FFD_, DIM_,
                DIM_, 1, 0, 0, 1,
                FFD_, 1, 0, 0, 1,
                FFD_, 1, 0, 0, 1,
                0, 0, 1.f, 1, 1);
    // x = x + gelu @ ff2_w + ff2_b
    launch_gemm(stream, X, QP, ff2_w + (size_t)i * FFD_ * DIM_,
                ff2_b + (size_t)i * DIM_, X,
                B_ * T_, DIM_, FFD_,
                FFD_, 1, 0, 0, 1,
                DIM_, 1, 0, 0, 1,
                DIM_, 1, 0, 0, 1,
                0, 0, 1.f, 2, 1);
  }

  mask_kernel<<<(B_ * NM_ * T_) / 256, 256, 0, stream>>>((float*)d_out, X, mask_w, mask_b);
}
